// LSTM_LM_7206955123404
// MI455X (gfx1250) — compile-verified, matrix operand order checked by reading, not run on hardware
//
#include <hip/hip_runtime.h>
#include <math.h>

// ---------------------------------------------------------------------------
// LSTM LM forward on gfx1250 (MI455X).
//   VOCAB=32000, EMB=HID=1024, NLAYERS=2, LIMIT=128, BATCH=32
// v4: FC kernel now stages its 64-row W_fc panel into LDS with the Tensor
//     Data Mover (tensor_load_to_lds + s_wait_tensorcnt), feeding WMMAs from
//     LDS instead of per-tile global loads. Layer kernels unchanged.
// ---------------------------------------------------------------------------

typedef __attribute__((ext_vector_type(8)))  _Float16 v8h;
typedef __attribute__((ext_vector_type(16))) _Float16 v16h;
typedef __attribute__((ext_vector_type(8)))  float    v8f;
typedef __attribute__((ext_vector_type(4)))  unsigned int u32x4;
typedef __attribute__((ext_vector_type(8)))  int      i32x8;
typedef __attribute__((ext_vector_type(4)))  int      i32x4;

#define VOCAB 32000
#define EMB   1024
#define HID   1024
#define LIMIT 128
#define BATCH 32

// Load one WMMA operand tile in the ISA 16-bit A layout (16x32, ld = 1024):
//   lane<16  : row = lane,    K = {kc*32+0..7,  kc*32+16..23}
//   lane>=16 : row = lane-16, K = {kc*32+8..15, kc*32+24..31}
__device__ __forceinline__ v16h load_tile(const _Float16* __restrict__ base,
                                          int row, int kb) {
  const _Float16* p = base + (size_t)row * 1024 + kb;
  v8h lo = *(const v8h*)(p);        // 16B aligned
  v8h hi = *(const v8h*)(p + 16);
  return __builtin_shufflevector(lo, hi, 0,1,2,3,4,5,6,7,8,9,10,11,12,13,14,15);
}

// Same tile shape, but from LDS (ds_load path).
__device__ __forceinline__ v16h lds_load_tile(const _Float16* base, int row, int kb) {
  const _Float16* p = base + (size_t)row * 1024 + kb;
  v8h lo = *(const v8h*)(p);
  v8h hi = *(const v8h*)(p + 16);
  return __builtin_shufflevector(lo, hi, 0,1,2,3,4,5,6,7,8,9,10,11,12,13,14,15);
}

__device__ __forceinline__ float sigm(float x) { return 1.0f / (1.0f + __expf(-x)); }

__device__ __forceinline__ v8f wmma16(v16h a, v16h b, v8f c) {
  return __builtin_amdgcn_wmma_f32_16x16x32_f16(false, a, false, b,
                                                (short)0, c, false, false);
}

// ---------------------------------------------------------------------------
// TDM: DMA a 2D tile (tile_d1 rows x tile_d0 elements, 2-byte data) from a
// row-major tensor (tensor_d1 x tensor_d0, stride0 elements) into LDS.
// D# packing per CDNA5 ISA ch.8 (group0 128b, group1 256b; groups 2/3 unused
// for 2D tensors).
// ---------------------------------------------------------------------------
__device__ __forceinline__ void tdm_load_2d(unsigned lds_byte_addr,
                                            const void* gaddr,
                                            unsigned tensor_d0, unsigned tensor_d1,
                                            unsigned tile_d0, unsigned tile_d1,
                                            unsigned stride0) {
  unsigned long long ga = (unsigned long long)(size_t)gaddr;
  u32x4 g0;
  g0.x = 1u;                                             // count=1, user D#
  g0.y = lds_byte_addr;                                  // LDS dest (bytes)
  g0.z = (unsigned)(ga & 0xFFFFFFFFull);                 // global_addr[31:0]
  g0.w = (unsigned)((ga >> 32) & 0x1FFFFFFull)           // global_addr[56:32]
         | (2u << 30);                                   // type = 2 ("image")
  i32x8 g1;
  g1[0] = (int)(1u << 16);                               // data_size=1 (2 bytes)
  g1[1] = (int)(tensor_d0 << 16);                        // tensor_dim0[15:0]
  g1[2] = (int)((tensor_d0 >> 16) | (tensor_d1 << 16));  // td0 hi | td1 lo
  g1[3] = (int)((tensor_d1 >> 16) | (tile_d0 << 16));    // td1 hi | tile_dim0
  g1[4] = (int)(tile_d1 & 0xFFFFu);                      // tile_dim1 (tile_dim2=0)
  g1[5] = (int)stride0;                                  // tensor_dim0_stride lo
  g1[6] = 0;                                             // stride0 hi | stride1 lo
  g1[7] = 0;                                             // stride1 hi
  i32x4 z4 = {0, 0, 0, 0};
#if defined(__clang_major__) && __clang_major__ >= 23
  i32x8 z8 = {0, 0, 0, 0, 0, 0, 0, 0};
  __builtin_amdgcn_tensor_load_to_lds(g0, g1, z4, z4, z8, 0);
#else
  __builtin_amdgcn_tensor_load_to_lds(g0, g1, z4, z4, 0);
#endif
}

// ---------------------------------------------------------------------------
// fp32 -> f16 weight conversion (grid-stride)
// ---------------------------------------------------------------------------
__global__ void cvt_kernel(const float* __restrict__ src, _Float16* __restrict__ dst,
                           size_t n) {
  size_t i = (size_t)blockIdx.x * blockDim.x + threadIdx.x;
  size_t stride = (size_t)gridDim.x * blockDim.x;
  for (; i < n; i += stride) dst[i] = (_Float16)src[i];
}

// ---------------------------------------------------------------------------
// init: zero h (buffer 0, both layers) and c; x16 = f16(emb[x[:,0]])
// ---------------------------------------------------------------------------
__global__ void init_kernel(const int* __restrict__ xids, const float* __restrict__ emb,
                            _Float16* __restrict__ x16, _Float16* __restrict__ hbuf0,
                            float* __restrict__ cst) {
  int tid = blockIdx.x * blockDim.x + threadIdx.x;   // 0..32767
  int b = tid >> 10;
  int k = tid & 1023;
  int tok = xids[b * (LIMIT + 1)];
  x16[tid] = (_Float16)emb[(size_t)tok * EMB + k];
  hbuf0[tid]             = (_Float16)0.0f;  // layer 0
  hbuf0[32 * 1024 + tid] = (_Float16)0.0f;  // layer 1
  cst[tid]               = 0.0f;
  cst[32 * 1024 + tid]   = 0.0f;
}

// ---------------------------------------------------------------------------
// Fused LSTM layer (unchanged from v3): split-K=2 + LDS reduction + fused
// cell epilogue; 2-deep register pipeline over K-chunks.
// Grid: 32 blocks x 256 threads.
// ---------------------------------------------------------------------------
#define LOADCHUNK(AX, AH, BT, KC)                                   \
  do {                                                              \
    const int kb_ = (KC) * 32 + 8 * hs;                             \
    AX = load_tile(in16,  arow, kb_);                               \
    AH = load_tile(hprev, arow, kb_);                               \
    _Pragma("unroll")                                               \
    for (int g_ = 0; g_ < 4; ++g_) {                                \
      BT[g_]     = load_tile(Wih, j0 + g_ * HID + r, kb_);          \
      BT[4 + g_] = load_tile(Whh, j0 + g_ * HID + r, kb_);          \
    }                                                               \
  } while (0)

#define DOWMMA(AX, AH, BT)                                          \
  do {                                                              \
    _Pragma("unroll")                                               \
    for (int g_ = 0; g_ < 4; ++g_) acc[g_] = wmma16(AX, BT[g_], acc[g_]); \
    _Pragma("unroll")                                               \
    for (int g_ = 0; g_ < 4; ++g_) acc[g_] = wmma16(AH, BT[4 + g_], acc[g_]); \
  } while (0)

__global__ __launch_bounds__(256) void lstm_layer_kernel(
    const _Float16* __restrict__ in16,
    const _Float16* __restrict__ hprev,
    _Float16*       __restrict__ hnext,
    float*          __restrict__ cst,
    const _Float16* __restrict__ Wih,
    const _Float16* __restrict__ Whh,
    const float*    __restrict__ bih,
    const float*    __restrict__ bhh)
{
  __shared__ float lds_part[4 * 4 * 256];           // 16KB

  const int lane     = threadIdx.x & 31;
  const int wv       = threadIdx.x >> 5;
  const int pairBlk  = wv >> 1;
  const int khalf    = wv & 1;
  const int grp      = blockIdx.x * 4 + pairBlk;    // 0..127
  const int mtile    = grp & 1;
  const int j0       = (grp >> 1) * 16;
  const int r        = lane & 15;
  const int hs       = lane >> 4;
  const int arow     = mtile * 16 + r;

  v8f acc[4] = {};
  const int kc0 = khalf * 16;

  v16h Ax0, Ah0, Bt0[8];
  v16h Ax1, Ah1, Bt1[8];
  LOADCHUNK(Ax0, Ah0, Bt0, kc0);
#pragma unroll
  for (int i = 0; i < 8; ++i) {
    LOADCHUNK(Ax1, Ah1, Bt1, kc0 + 2 * i + 1);
    DOWMMA(Ax0, Ah0, Bt0);
    if (i < 7)
      LOADCHUNK(Ax0, Ah0, Bt0, kc0 + 2 * i + 2);
    DOWMMA(Ax1, Ah1, Bt1);
  }

  if (khalf == 1) {
#pragma unroll
    for (int g = 0; g < 4; ++g)
#pragma unroll
      for (int v = 0; v < 8; ++v)
        lds_part[(pairBlk * 4 + g) * 256 + lane * 8 + v] = acc[g][v];
  }
  __syncthreads();
  if (khalf == 1) return;
#pragma unroll
  for (int g = 0; g < 4; ++g)
#pragma unroll
    for (int v = 0; v < 8; ++v)
      acc[g][v] += lds_part[(pairBlk * 4 + g) * 256 + lane * 8 + v];

  const int n = j0 + r;
  const float bi = bih[n]           + bhh[n];
  const float bf = bih[n + HID]     + bhh[n + HID];
  const float bg = bih[n + 2 * HID] + bhh[n + 2 * HID];
  const float bo = bih[n + 3 * HID] + bhh[n + 3 * HID];
#pragma unroll
  for (int v = 0; v < 8; ++v) {
    const int b = mtile * 16 + v + 8 * hs;
    const float ig = sigm(acc[0][v] + bi);
    const float fg = sigm(acc[1][v] + bf);
    const float gg = tanhf(acc[2][v] + bg);
    const float og = sigm(acc[3][v] + bo);
    const size_t idx = (size_t)b * HID + n;
    const float cn = fg * cst[idx] + ig * gg;
    cst[idx] = cn;
    hnext[idx] = (_Float16)(og * tanhf(cn));
  }
}

// ---------------------------------------------------------------------------
// FC: logits(32,32000) = h1(32,1024) @ Wfc^T + bfc.
// TDM stages a 64-row x 1024-col f16 panel of Wfc (128KB) into dynamic LDS;
// the K-loop runs ds_load + WMMA (no per-tile global weight loads).
// Grid: 500 blocks x 256 threads (8 waves = 2 M-tiles x 4 N-tiles),
// dynamic LDS = 131072 bytes.
// ---------------------------------------------------------------------------
__global__ __launch_bounds__(256) void fc_kernel(
    const _Float16* __restrict__ h16,     // (32,1024)
    const _Float16* __restrict__ Wfc,     // (32000,1024) f16
    const float*    __restrict__ bfc,     // (32000)
    float*          __restrict__ logits)  // (32,32000)
{
  extern __shared__ _Float16 Bpanel[];    // 64 rows x 1024 halves = 128KB

  const int lane  = threadIdx.x & 31;
  const int wv    = threadIdx.x >> 5;     // 0..7
  const int mtile = wv & 1;
  const int ntile = wv >> 1;              // 0..3
  const int nb0   = blockIdx.x * 64;      // panel row base in Wfc
  const int r     = lane & 15;
  const int hs    = lane >> 4;
  const int arow  = mtile * 16 + r;

  // one TDM DMA for the whole panel, issued by wave 0 only
  if (wv == 0) {
    tdm_load_2d((unsigned)(size_t)&Bpanel[0],
                Wfc + (size_t)nb0 * 1024,
                /*tensor_d0=*/1024, /*tensor_d1=*/VOCAB,
                /*tile_d0=*/1024,  /*tile_d1=*/64,
                /*stride0=*/1024);
    __builtin_amdgcn_s_wait_tensorcnt(0);
  }
  __syncthreads();

  v8f acc = {};
  const int brow = ntile * 16 + r;        // row within LDS panel
#pragma unroll 4
  for (int kc = 0; kc < 32; ++kc) {
    const int kb = kc * 32 + 8 * hs;
    v16h A = load_tile(h16, arow, kb);          // global, cache-hot (64KB)
    v16h B = lds_load_tile(Bpanel, brow, kb);   // LDS
    acc = wmma16(A, B, acc);
  }

  const int n = nb0 + ntile * 16 + r;
  const float bias = bfc[n];
#pragma unroll
  for (int v = 0; v < 8; ++v) {
    const int b = mtile * 16 + v + 8 * hs;
    logits[(size_t)b * VOCAB + n] = acc[v] + bias;
  }
}

// ---------------------------------------------------------------------------
// Finalize: per batch row -> max/argmax, sum(exp), p(target), next input row.
// Grid: 32 blocks x 512 threads.
// ---------------------------------------------------------------------------
__global__ __launch_bounds__(512) void finalize_kernel(
    const float* __restrict__ logits,
    const int*   __restrict__ xids,
    const float* __restrict__ emb,
    _Float16*    __restrict__ xnext16,
    float*       __restrict__ out,
    int t)
{
  __shared__ float smax[512];
  __shared__ int   sidx[512];
  __shared__ float ssum[512];
  __shared__ float sM;
  __shared__ int   sPred;

  const int b = blockIdx.x;
  const int tid = threadIdx.x;
  const float* row = logits + (size_t)b * VOCAB;

  float m = -3.4e38f; int mi = 0;
  for (int j = tid; j < VOCAB; j += 512) {
    float v = row[j];
    if (v > m) { m = v; mi = j; }
  }
  smax[tid] = m; sidx[tid] = mi; __syncthreads();
  for (int s = 256; s > 0; s >>= 1) {
    if (tid < s) {
      float a = smax[tid], c = smax[tid + s];
      if (c > a || (c == a && sidx[tid + s] < sidx[tid])) {
        smax[tid] = c; sidx[tid] = sidx[tid + s];
      }
    }
    __syncthreads();
  }
  if (tid == 0) { sM = smax[0]; sPred = sidx[0]; }
  __syncthreads();
  const float M = sM;

  float s = 0.0f;
  for (int j = tid; j < VOCAB; j += 512) s += __expf(row[j] - M);
  ssum[tid] = s; __syncthreads();
  for (int st = 256; st > 0; st >>= 1) {
    if (tid < st) ssum[tid] += ssum[tid + st];
    __syncthreads();
  }

  if (tid == 0) {
    const int tgt = xids[b * (LIMIT + 1) + t + 1];
    out[b * LIMIT + t] = __expf(row[tgt] - M) / ssum[0];
  }

  const int pred = sPred;
  const float* erow = emb + (size_t)pred * EMB;
  for (int k = tid; k < EMB; k += 512)
    xnext16[(size_t)b * EMB + k] = (_Float16)erow[k];
}

// ---------------------------------------------------------------------------
// Host launcher
// ---------------------------------------------------------------------------
extern "C" void kernel_launch(void* const* d_in, const int* in_sizes, int n_in,
                              void* d_out, int out_size, void* d_ws, size_t ws_size,
                              hipStream_t stream) {
  (void)in_sizes; (void)n_in; (void)out_size; (void)ws_size;

  const int*   xids = (const int*)  d_in[0];
  const float* emb  = (const float*)d_in[1];
  const float* Wih  = (const float*)d_in[2];
  const float* Whh  = (const float*)d_in[3];
  const float* bih  = (const float*)d_in[4];
  const float* bhh  = (const float*)d_in[5];
  const float* Wfc  = (const float*)d_in[6];
  const float* bfc  = (const float*)d_in[7];
  float* out = (float*)d_out;

  char* ws = (char*)d_ws;
  size_t off = 0;
  auto carve = [&](size_t bytes) -> char* {
    char* p = ws + off;
    off += (bytes + 255) & ~(size_t)255;
    return p;
  };
  _Float16* Wih16  = (_Float16*)carve(2ull * 4096 * 1024 * sizeof(_Float16));
  _Float16* Whh16  = (_Float16*)carve(2ull * 4096 * 1024 * sizeof(_Float16));
  _Float16* Wfc16  = (_Float16*)carve((size_t)VOCAB * 1024 * sizeof(_Float16));
  _Float16* x16    = (_Float16*)carve((size_t)BATCH * EMB * sizeof(_Float16));
  _Float16* hbuf   = (_Float16*)carve(2ull * 2 * BATCH * HID * sizeof(_Float16));
  float*    cst    = (float*)   carve(2ull * BATCH * HID * sizeof(float));
  float*    logits = (float*)   carve((size_t)BATCH * VOCAB * sizeof(float));

  cvt_kernel<<<2048, 256, 0, stream>>>(Wih, Wih16, 2ull * 4096 * 1024);
  cvt_kernel<<<2048, 256, 0, stream>>>(Whh, Whh16, 2ull * 4096 * 1024);
  cvt_kernel<<<4096, 256, 0, stream>>>(Wfc, Wfc16, (size_t)VOCAB * 1024);
  init_kernel<<<128, 256, 0, stream>>>(xids, emb, x16, hbuf, cst);

  const size_t HBUF = 2ull * BATCH * HID;
  const size_t LOFF = (size_t)BATCH * HID;

  for (int t = 0; t < LIMIT; ++t) {
    const int cur = t & 1, nxt = cur ^ 1;
    _Float16* hc = hbuf + (size_t)cur * HBUF;
    _Float16* hn = hbuf + (size_t)nxt * HBUF;

    lstm_layer_kernel<<<32, 256, 0, stream>>>(
        x16, hc, hn, cst, Wih16, Whh16, bih, bhh);
    lstm_layer_kernel<<<32, 256, 0, stream>>>(
        hn, hc + LOFF, hn + LOFF, cst + LOFF,
        Wih16 + 4096ull * 1024, Whh16 + 4096ull * 1024, bih + 4096, bhh + 4096);
    fc_kernel<<<500, 256, 131072, stream>>>(hn + LOFF, Wfc16, bfc, logits);
    finalize_kernel<<<32, 512, 0, stream>>>(logits, xids, emb, x16, out, t);
  }
}
